// LocalLoraAttention_3796751090124
// MI455X (gfx1250) — matrix operand order repaired and checked by reading, not compiled, not run on hardware
//
#include <hip/hip_runtime.h>

// ---------------------------------------------------------------- constants
#define Bc   2
#define Sc   2048
#define Hc   2048
#define NHc  16
#define HDc  128
#define Rc   128
#define Mc   (Bc * Sc)          // 4096 tokens
#define LORA_SCALE 2.0f         // 256/128
#define NEGF (-1.0e30f)

typedef __attribute__((ext_vector_type(16))) __bf16 v16bf;
typedef __attribute__((ext_vector_type(8)))  float  v8f;

union BF16Frag {                // 16 bf16 = 32 bytes
    v16bf v;
    uint4 u[2];
};

__device__ __forceinline__ v8f wmma_bf16(const v16bf& a, const v16bf& b, const v8f& c) {
    return __builtin_amdgcn_wmma_f32_16x16x32_bf16(false, a, false, b, (short)0, c,
                                                   false, false);
}

// ------------------------------------------------- async global->LDS (CDNA5)
// LDS flat pointers: low 32 bits are the LDS offset (shared aperture mapping).
__device__ __forceinline__ void async_copy16(void* lds_dst, const void* gsrc) {
    unsigned           la = (unsigned)(size_t)lds_dst;
    unsigned long long ga = (unsigned long long)(size_t)gsrc;
    asm volatile("global_load_async_to_lds_b128 %0, %1, off"
                 :: "v"(la), "v"(ga) : "memory");
}
template <int N_> __device__ __forceinline__ void wait_async() {
    asm volatile("s_wait_asynccnt %0" :: "i"(N_) : "memory");
}

// ---------------------------------------------------------------- GEMM
// C[M,N] (f32) =/+= A[M,K] (bf16, row-major) x WT[K,N] (bf16, row-major).
// Double-buffered GLOBAL_LOAD_ASYNC_TO_LDS_B128 pipeline: tile t+1 DMA runs
// under tile t's WMMAs; ASYNCcnt completes in-order so waiting <=4 releases
// the previous 4-chunk batch.
#define GT_M 128
#define GT_N 128
#define GT_K 32
#define A_PITCH 40              // 32 + 8 pad  (row = 80B, 16B aligned)
#define W_PITCH 136             // 128 + 8 pad (row = 272B, 16B aligned)

__global__ __launch_bounds__(256) void gemm_bf16_wmma(
    float* __restrict__ C, const __bf16* __restrict__ A, const __bf16* __restrict__ WT,
    int M, int N, int K, int accumulate)
{
    __shared__ __bf16 At[2][GT_M * A_PITCH];   // [m][k]
    __shared__ __bf16 Bt[2][GT_K * W_PITCH];   // [k][n]

    const int tid  = threadIdx.x;
    const int lane = tid & 31;
    const int wave = tid >> 5;
    const int wm   = wave >> 1;             // 0..3 -> 32-row slab
    const int wn   = wave & 1;              // 0..1 -> 64-col slab
    const int bm   = blockIdx.y * GT_M;
    const int bn   = blockIdx.x * GT_N;
    const int mhalf = lane >> 4;
    const int ncol  = lane & 15;

    // per-thread staging geometry: 2 A-chunks + 2 B-chunks of 16B each
    const int ar0 = tid >> 2,          aq0 = (tid & 3) * 8;
    const int ar1 = ar0 + 64,          aq1 = aq0;
    const int br0 = tid >> 4,          bq0 = (tid & 15) * 8;
    const int br1 = br0 + 16,          bq1 = bq0;
    const __bf16* pa0 = A  + (size_t)(bm + ar0) * K + aq0;
    const __bf16* pa1 = A  + (size_t)(bm + ar1) * K + aq1;
    const __bf16* pb0 = WT + (size_t)br0 * N + bn + bq0;
    const __bf16* pb1 = WT + (size_t)br1 * N + bn + bq1;
    const size_t  stepB = (size_t)GT_K * N;

    auto stage = [&](int buf) {
        async_copy16(&At[buf][ar0 * A_PITCH + aq0], pa0);
        async_copy16(&At[buf][ar1 * A_PITCH + aq1], pa1);
        async_copy16(&Bt[buf][br0 * W_PITCH + bq0], pb0);
        async_copy16(&Bt[buf][br1 * W_PITCH + bq1], pb1);
        pa0 += GT_K; pa1 += GT_K; pb0 += stepB; pb1 += stepB;
    };

    v8f acc[2][4];
    if (accumulate) {
        for (int mt = 0; mt < 2; ++mt)
            for (int r = 0; r < 8; ++r) {
                int row = bm + wm * 32 + mt * 16 + r + 8 * mhalf;
                const float* rp = C + (size_t)row * N + bn + wn * 64 + ncol;
                for (int nt = 0; nt < 4; ++nt) acc[mt][nt][r] = rp[nt * 16];
            }
    } else {
        for (int mt = 0; mt < 2; ++mt)
            for (int nt = 0; nt < 4; ++nt)
                for (int r = 0; r < 8; ++r) acc[mt][nt][r] = 0.0f;
    }

    const int nk = K / GT_K;
    stage(0);
    for (int t = 0; t < nk; ++t) {
        const int buf = t & 1;
        if (t + 1 < nk) { stage(buf ^ 1); wait_async<4>(); }
        else            { wait_async<0>(); }
        __syncthreads();                       // all waves' tile-t writes visible

        BF16Frag afr[2], bfr[4];
        for (int mt = 0; mt < 2; ++mt) {
            int row = wm * 32 + mt * 16 + ncol;
            int o   = mhalf * 8;
            afr[mt].u[0] = *(const uint4*)(&At[buf][row * A_PITCH + o]);
            afr[mt].u[1] = *(const uint4*)(&At[buf][row * A_PITCH + o + 16]);
        }
        for (int nt = 0; nt < 4; ++nt) {
            const __bf16* p = &Bt[buf][(mhalf * 8) * W_PITCH + wn * 64 + nt * 16 + ncol];
            for (int s = 0; s < 8; ++s) {
                bfr[nt].v[s]     = p[s * W_PITCH];
                bfr[nt].v[8 + s] = p[(16 + s) * W_PITCH];
            }
        }
        for (int mt = 0; mt < 2; ++mt)
            for (int nt = 0; nt < 4; ++nt)
                acc[mt][nt] = wmma_bf16(afr[mt].v, bfr[nt].v, acc[mt][nt]);
        __syncthreads();                       // reads done before buf is re-staged
    }

    for (int mt = 0; mt < 2; ++mt)
        for (int r = 0; r < 8; ++r) {
            int row = bm + wm * 32 + mt * 16 + r + 8 * mhalf;
            float* rp = C + (size_t)row * N + bn + wn * 64 + ncol;
            for (int nt = 0; nt < 4; ++nt) rp[nt * 16] = acc[mt][nt][r];
        }
}

// ------------------------------------------------- converts / transposes
__global__ void cvt_kernel(const float* __restrict__ in, __bf16* __restrict__ out, int n)
{
    int idx = blockIdx.x * blockDim.x + threadIdx.x;
    if (idx < n) out[idx] = (__bf16)in[idx];
}

// out[k*Nr + n] = (bf16) in[n*Kc + k]   (in: [Nr,Kc] f32 -> out: [Kc,Nr] bf16)
__global__ __launch_bounds__(256) void transpose_cvt_kernel(
    const float* __restrict__ in, __bf16* __restrict__ out, int Nr, int Kc)
{
    __shared__ float tile[32][33];
    int bk = blockIdx.x * 32;
    int bn = blockIdx.y * 32;
    for (int i = 0; i < 32; i += 8) {
        int n = bn + threadIdx.y + i;
        int k = bk + threadIdx.x;
        tile[threadIdx.y + i][threadIdx.x] = in[(size_t)n * Kc + k];
    }
    __syncthreads();
    for (int i = 0; i < 32; i += 8) {
        int k = bk + threadIdx.y + i;
        int n = bn + threadIdx.x;
        out[(size_t)k * Nr + n] = (__bf16)tile[threadIdx.x][threadIdx.y + i];
    }
}

// lora mid: tb = bf16( t * scale * mask[token] ),  R = 128
__global__ void mask_scale_cvt_kernel(const float* __restrict__ T,
                                      const float* __restrict__ mask,
                                      __bf16* __restrict__ O, float scale, int n)
{
    int idx = blockIdx.x * blockDim.x + threadIdx.x;
    if (idx < n) O[idx] = (__bf16)(T[idx] * (scale * mask[idx >> 7]));
}

// ---------------------------------------------------------------- RoPE
// Q: f32 [B,S,NH,HD] -> bf16 [B,NH,S,HD]
__global__ void rope_cvt_q_kernel(const float* __restrict__ X, __bf16* __restrict__ Y)
{
    int idx = blockIdx.x * blockDim.x + threadIdx.x;   // B*NH*S*64
    int i = idx & 63;
    int s = (idx >> 6) & (Sc - 1);
    int h = (idx >> 17) & (NHc - 1);
    int b = idx >> 21;
    const float* xp = X + (((size_t)(b * Sc + s)) * NHc + h) * HDc;
    float x1 = xp[i], x2 = xp[i + 64];
    float invf = __powf(10000.0f, -(2.0f * (float)i) * (1.0f / 128.0f));
    float sn, cs;
    __sincosf((float)s * invf, &sn, &cs);
    __bf16* yp = Y + (((size_t)(b * NHc + h)) * Sc + s) * HDc;
    yp[i]      = (__bf16)(x1 * cs - x2 * sn);
    yp[i + 64] = (__bf16)(x2 * cs + x1 * sn);
}

// K: f32 [B,S,NH,HD] -> bf16 [B,NH,HD,S]  (pre-transposed for QK^T B-fragments)
__global__ void rope_cvt_kT_kernel(const float* __restrict__ X, __bf16* __restrict__ Y)
{
    int idx = blockIdx.x * blockDim.x + threadIdx.x;
    int i = idx & 63;
    int s = (idx >> 6) & (Sc - 1);
    int h = (idx >> 17) & (NHc - 1);
    int b = idx >> 21;
    const float* xp = X + (((size_t)(b * Sc + s)) * NHc + h) * HDc;
    float x1 = xp[i], x2 = xp[i + 64];
    float invf = __powf(10000.0f, -(2.0f * (float)i) * (1.0f / 128.0f));
    float sn, cs;
    __sincosf((float)s * invf, &sn, &cs);
    __bf16* yp = Y + ((size_t)(b * NHc + h)) * HDc * Sc;
    yp[(size_t)i * Sc + s]        = (__bf16)(x1 * cs - x2 * sn);
    yp[(size_t)(i + 64) * Sc + s] = (__bf16)(x2 * cs + x1 * sn);
}

// V: f32 [B,S,NH,HD] -> bf16 [B,NH,S,HD]
__global__ void vcvt_kernel(const float* __restrict__ X, __bf16* __restrict__ Y)
{
    int idx = blockIdx.x * blockDim.x + threadIdx.x;
    int d = idx & 127;
    int s = (idx >> 7) & (Sc - 1);
    int h = (idx >> 18) & (NHc - 1);
    int b = idx >> 22;
    Y[(((size_t)(b * NHc + h)) * Sc + s) * HDc + d] =
        (__bf16)X[(((size_t)(b * Sc + s)) * NHc + h) * HDc + d];
}

// ---------------------------------------------------------------- flash attn
// Q bf16 [B,NH,S,HD]; KT bf16 [B,NH,HD,S]; V bf16 [B,NH,S,HD];
// Out bf16 [B,S,H]. Causal, online softmax; double-buffered async KV staging.
__global__ __launch_bounds__(128) void flash_attn_kernel(
    const __bf16* __restrict__ Qb, const __bf16* __restrict__ KTb,
    const __bf16* __restrict__ Vb, __bf16* __restrict__ Out)
{
    __shared__ __bf16 Kt[2][HDc * A_PITCH];  // [hd][kv], pitch 40
    __shared__ __bf16 Vt[2][32 * W_PITCH];   // [kv][d],  pitch 136
    __shared__ __bf16 Pl[4 * 16 * A_PITCH];  // per-wave P staging [16][40]

    const int tid  = threadIdx.x;
    const int lane = tid & 31;
    const int wave = tid >> 5;
    const int qblk = blockIdx.x;
    const int h    = blockIdx.y;
    const int b    = blockIdx.z;
    const size_t headoff = ((size_t)(b * NHc + h)) * Sc * HDc;  // same for Q/KT/V
    const int qbase = qblk * 64 + wave * 16;
    const int mhalf = lane >> 4;
    const int ncol  = lane & 15;
    const int mrow  = 8 * mhalf;
    const float scl = 0.08838834764831845f;  // 1/sqrt(128)

    // staging geometry: per thread 4 K-chunks + 4 V-chunks of 16B per step
    int ko[4], kl[4], vo[4], vl[4];
    for (int i = 0; i < 4; ++i) {
        int c  = tid + i * 128;
        int hd = c >> 2,  q  = (c & 3) * 8;
        ko[i] = hd * Sc + q;
        kl[i] = hd * A_PITCH + q;
        int kv = c >> 4,  dc = (c & 15) * 8;
        vo[i] = kv * HDc + dc;
        vl[i] = kv * W_PITCH + dc;
    }
    const __bf16* kBase = KTb + headoff;
    const __bf16* vBase = Vb  + headoff;

    auto stage = [&](int buf) {
        for (int i = 0; i < 4; ++i) {
            async_copy16(&Kt[buf][kl[i]], kBase + ko[i]);
            async_copy16(&Vt[buf][vl[i]], vBase + vo[i]);
        }
        kBase += 32;            // next 32 kv columns of KT rows
        vBase += 32 * HDc;      // next 32 kv rows of V
    };

    // Q fragments (4 K-chunks of 32 over HD), kept in registers
    BF16Frag qf[4];
    {
        const __bf16* qp = Qb + headoff + (size_t)(qbase + ncol) * HDc;
        int o = mhalf * 8;
        for (int kk = 0; kk < 4; ++kk) {
            qf[kk].u[0] = *(const uint4*)(qp + kk * 32 + o);
            qf[kk].u[1] = *(const uint4*)(qp + kk * 32 + 16 + o);
        }
    }

    v8f o_acc[8];
    float m_i[8], l_i[8];
    for (int j = 0; j < 8; ++j)
        for (int r = 0; r < 8; ++r) o_acc[j][r] = 0.0f;
    for (int r = 0; r < 8; ++r) { m_i[r] = NEGF; l_i[r] = 0.0f; }

    const int nsteps = qblk * 2 + 2;         // kv covers [0, qblk*64+64)
    stage(0);
    for (int stp = 0; stp < nsteps; ++stp) {
        const int kvb = stp * 32;
        const int buf = stp & 1;
        if (stp + 1 < nsteps) { stage(buf ^ 1); wait_async<8>(); }
        else                  { wait_async<0>(); }
        __syncthreads();                     // step-stp K/V tiles visible

        // scores: two 16x16 key tiles
        v8f s0, s1;
        for (int r = 0; r < 8; ++r) { s0[r] = 0.0f; s1[r] = 0.0f; }
        for (int kk = 0; kk < 4; ++kk) {
            BF16Frag b0, b1;
            const __bf16* kp = &Kt[buf][(kk * 32 + mhalf * 8) * A_PITCH + ncol];
            for (int s = 0; s < 8; ++s) {
                b0.v[s] = kp[s * A_PITCH];      b0.v[8 + s] = kp[(16 + s) * A_PITCH];
                b1.v[s] = kp[s * A_PITCH + 16]; b1.v[8 + s] = kp[(16 + s) * A_PITCH + 16];
            }
            s0 = wmma_bf16(qf[kk].v, b0.v, s0);
            s1 = wmma_bf16(qf[kk].v, b1.v, s1);
        }

        // online softmax (rows live in 16-lane half-groups)
        float p0[8], p1[8], corr[8];
        const int key0 = kvb + ncol;
        for (int r = 0; r < 8; ++r) {
            int qg = qbase + mrow + r;
            float a0 = s0[r] * scl; if (key0 > qg)      a0 = NEGF;
            float a1 = s1[r] * scl; if (key0 + 16 > qg) a1 = NEGF;
            float rm = fmaxf(a0, a1);
            rm = fmaxf(rm, __shfl_xor(rm, 1, 32));
            rm = fmaxf(rm, __shfl_xor(rm, 2, 32));
            rm = fmaxf(rm, __shfl_xor(rm, 4, 32));
            rm = fmaxf(rm, __shfl_xor(rm, 8, 32));
            float mn = fmaxf(m_i[r], rm);
            corr[r] = __expf(m_i[r] - mn);
            m_i[r]  = mn;
            p0[r] = __expf(a0 - mn);
            p1[r] = __expf(a1 - mn);
            float rs = p0[r] + p1[r];
            rs += __shfl_xor(rs, 1, 32);
            rs += __shfl_xor(rs, 2, 32);
            rs += __shfl_xor(rs, 4, 32);
            rs += __shfl_xor(rs, 8, 32);
            l_i[r] = l_i[r] * corr[r] + rs;
        }
        for (int j = 0; j < 8; ++j)
            for (int r = 0; r < 8; ++r) o_acc[j][r] *= corr[r];

        // C-layout P -> LDS -> A-fragment layout
        for (int r = 0; r < 8; ++r) {
            Pl[(wave * 16 + mrow + r) * A_PITCH + ncol]      = (__bf16)p0[r];
            Pl[(wave * 16 + mrow + r) * A_PITCH + 16 + ncol] = (__bf16)p1[r];
        }
        __syncthreads();
        BF16Frag pf;
        {
            const __bf16* pp = &Pl[(wave * 16 + ncol) * A_PITCH + mhalf * 8];
            pf.u[0] = *(const uint4*)(pp);
            pf.u[1] = *(const uint4*)(pp + 16);
        }
        // O += P x V  (8 d-tiles)
        for (int j = 0; j < 8; ++j) {
            BF16Frag vf;
            const __bf16* vp = &Vt[buf][(mhalf * 8) * W_PITCH + j * 16 + ncol];
            for (int s = 0; s < 8; ++s) {
                vf.v[s]     = vp[s * W_PITCH];
                vf.v[8 + s] = vp[(16 + s) * W_PITCH];
            }
            o_acc[j] = wmma_bf16(pf.v, vf.v, o_acc[j]);
        }
        __syncthreads();                     // reads done before buf re-staged
    }

    // epilogue: normalize, emit bf16 [B,S,H] (feeds o-projection GEMM)
    for (int r = 0; r < 8; ++r) {
        float inv = 1.0f / l_i[r];
        int qg = qbase + mrow + r;
        __bf16* op = Out + ((size_t)b * Sc + qg) * Hc + h * HDc + ncol;
        for (int j = 0; j < 8; ++j) op[j * 16] = (__bf16)(o_acc[j][r] * inv);
    }
}

// ---------------------------------------------------------------- launch
extern "C" void kernel_launch(void* const* d_in, const int* in_sizes, int n_in,
                              void* d_out, int out_size, void* d_ws, size_t ws_size,
                              hipStream_t stream)
{
    const float* x  = (const float*)d_in[0];
    const float* md = (const float*)d_in[1];
    const float* mv = (const float*)d_in[2];

    char* w = (char*)d_ws;
    float*  qf   = (float*)w;  w += (size_t)Mc * Hc * 4;    // 33.5 MB
    float*  kf   = (float*)w;  w += (size_t)Mc * Hc * 4;
    float*  vf   = (float*)w;  w += (size_t)Mc * Hc * 4;
    float*  tf   = (float*)w;  w += (size_t)Mc * Rc * 4;    // 2 MB lora mid (f32)
    __bf16* xb   = (__bf16*)w; w += (size_t)Mc * Hc * 2;    // bf16 activations
    __bf16* tb   = (__bf16*)w; w += (size_t)Mc * Rc * 2;    // bf16 lora mid
    __bf16* WTb  = (__bf16*)w; w += (size_t)Hc * Hc * 2;    // transposed weight
    __bf16* ATb  = (__bf16*)w; w += (size_t)Hc * Rc * 2;    // transposed lora-A
    __bf16* BTb  = (__bf16*)w; w += (size_t)Rc * Hc * 2;    // transposed lora-B
    __bf16* qb2  = (__bf16*)w; w += (size_t)Mc * Hc * 2;    // [B,NH,S,HD]
    __bf16* kT2  = (__bf16*)w; w += (size_t)Mc * Hc * 2;    // [B,NH,HD,S]
    __bf16* vb2  = (__bf16*)w; w += (size_t)Mc * Hc * 2;    // [B,NH,S,HD]
    __bf16* aob  = (__bf16*)qf;  // qf dead after rope; reuse for attn output bf16

    dim3 gBig(Hc / GT_N, Mc / GT_M);   // (16, 32)
    dim3 gLoraA(Rc / GT_N, Mc / GT_M); // (1, 32)
    dim3 tb32(32, 8);

    cvt_kernel<<<(Mc * Hc + 255) / 256, 256, 0, stream>>>(x, xb, Mc * Hc);

    // fused base + LoRA projection: out = Xb WT + s*(md * Xb A_d^T) B_d^T + ...
    auto proj = [&](const __bf16* Xb, int wi, float* out) {
        transpose_cvt_kernel<<<dim3(Hc / 32, Hc / 32), tb32, 0, stream>>>(
            (const float*)d_in[wi], WTb, Hc, Hc);
        gemm_bf16_wmma<<<gBig, 256, 0, stream>>>(out, Xb, WTb, Mc, Hc, Hc, 0);
        for (int ad = 0; ad < 2; ++ad) {
            const float* Aw = (const float*)d_in[wi + 1 + 2 * ad];  // [R,H]
            const float* Bw = (const float*)d_in[wi + 2 + 2 * ad];  // [H,R]
            const float* mk = (ad == 0) ? md : mv;
            transpose_cvt_kernel<<<dim3(Hc / 32, Rc / 32), tb32, 0, stream>>>(
                Aw, ATb, Rc, Hc);                                   // -> [H,R]
            transpose_cvt_kernel<<<dim3(Rc / 32, Hc / 32), tb32, 0, stream>>>(
                Bw, BTb, Hc, Rc);                                   // -> [R,H]
            gemm_bf16_wmma<<<gLoraA, 256, 0, stream>>>(tf, Xb, ATb, Mc, Rc, Hc, 0);
            mask_scale_cvt_kernel<<<(Mc * Rc + 255) / 256, 256, 0, stream>>>(
                tf, mk, tb, LORA_SCALE, Mc * Rc);
            gemm_bf16_wmma<<<gBig, 256, 0, stream>>>(out, tb, BTb, Mc, Hc, Rc, 1);
        }
    };

    proj(xb, 3,  qf);   // Wq + LoRA
    proj(xb, 8,  kf);   // Wk + LoRA
    proj(xb, 13, vf);   // Wv + LoRA

    int nr = Bc * NHc * Sc * 64;
    rope_cvt_q_kernel<<<nr / 256, 256, 0, stream>>>(qf, qb2);
    rope_cvt_kT_kernel<<<nr / 256, 256, 0, stream>>>(kf, kT2);
    int nv = Bc * NHc * Sc * HDc;
    vcvt_kernel<<<nv / 256, 256, 0, stream>>>(vf, vb2);

    flash_attn_kernel<<<dim3(Sc / 64, NHc, Bc), 128, 0, stream>>>(qb2, kT2, vb2, aob);

    proj(aob, 18, (float*)d_out);   // Wo + LoRA -> final f32 output
}
